// MetaNet_26130581029260
// MI455X (gfx1250) — matrix operand order repaired and checked by reading, not compiled, not run on hardware
//
#include <hip/hip_runtime.h>

typedef __attribute__((ext_vector_type(16))) __bf16 v16bf;
typedef __attribute__((ext_vector_type(8)))  __bf16 v8bf;
typedef __attribute__((ext_vector_type(4)))  __bf16 v4bf;
typedef __attribute__((ext_vector_type(8)))  float  v8f;

#define NUM_HEADS 14
#define P_MAX     36928
#define B_ROWS    1024
#define D_IN      1920
#define D_SLICE   128
#define N_HID     (NUM_HEADS * D_SLICE)   // 1792

// Column-tile prefix sums (p/16 per head) and head sizes. All SIZES % 16 == 0.
__constant__ int c_cto[NUM_HEADS + 1] = {0, 580, 1736, 4044, 6352, 8660, 10968, 13276,
                                         15584, 17892, 19048, 19628, 20784, 21940, 22520};
__constant__ int c_p[NUM_HEADS] = {9280, 18496, 36928, 36928, 36928, 36928, 36928,
                                   36928, 36928, 18496, 9280, 18496, 18496, 9280};

__device__ __forceinline__ v16bf cat16(v8bf lo, v8bf hi) {
    return __builtin_shufflevector(lo, hi, 0, 1, 2, 3, 4, 5, 6, 7,
                                           8, 9, 10, 11, 12, 13, 14, 15);
}

__device__ __forceinline__ v8bf cvt8f(const float* __restrict__ p) {
    const float4 f0 = *(const float4*)p;
    const float4 f1 = *(const float4*)(p + 4);
    v8bf r;
    r[0] = (__bf16)f0.x; r[1] = (__bf16)f0.y; r[2] = (__bf16)f0.z; r[3] = (__bf16)f0.w;
    r[4] = (__bf16)f1.x; r[5] = (__bf16)f1.y; r[6] = (__bf16)f1.z; r[7] = (__bf16)f1.w;
    return r;
}

// ---------------- f32 -> bf16 bulk convert (one float4 per thread) ----------------
__global__ void __launch_bounds__(256)
cvt_f32_bf16_kernel(const float* __restrict__ in, unsigned short* __restrict__ out, long n4) {
    long v = (long)blockIdx.x * blockDim.x + threadIdx.x;
    if (v >= n4) return;
    float4 f = ((const float4*)in)[v];
    v4bf r;
    r[0] = (__bf16)f.x; r[1] = (__bf16)f.y; r[2] = (__bf16)f.z; r[3] = (__bf16)f.w;
    *(v4bf*)((__bf16*)out + v * 4) = r;
}

// ---------------- Stage 1: hidden = relu(x @ Wh^T + b), bf16 out ----------------
// grid (14, 64), block 256. Wave w of block bx handles col tile bx*8+w (0..111),
// row tile = blockIdx.y (0..63). K = 1920, 60 WMMA steps.
template <bool BF16IN>
__global__ void __launch_bounds__(256)
stage1_kernel(const float* __restrict__ xf, const float* __restrict__ Whf,
              const unsigned short* __restrict__ xb, const unsigned short* __restrict__ Whb,
              const float* __restrict__ bh, unsigned short* __restrict__ hidden) {
    const int lane = threadIdx.x & 31;
    const int wave = threadIdx.x >> 5;
    const int l16  = lane & 15;
    const int hi   = lane >> 4;
    const int colTile = blockIdx.x * 8 + wave;   // 0..111
    const int rowTile = blockIdx.y;              // 0..63
    const int arow = rowTile * 16 + l16;         // x row
    const int brow = colTile * 16 + l16;         // W_hidden row (= hidden col)

    const size_t aoff = (size_t)arow * D_IN + hi * 8;   // + k, +16 for second chunk
    const size_t boff = (size_t)brow * D_IN + hi * 8;   // + k, +16 for second chunk

    v8f acc = {};
    for (int k = 0; k < D_IN; k += 32) {
        v16bf a, b;
        if constexpr (BF16IN) {
            const __bf16* ap = (const __bf16*)xb  + aoff + k;
            const __bf16* bp = (const __bf16*)Whb + boff + k;
            a = cat16(*(const v8bf*)ap, *(const v8bf*)(ap + 16));
            b = cat16(*(const v8bf*)bp, *(const v8bf*)(bp + 16));
        } else {
            const float* ap = xf  + aoff + k;
            const float* bp = Whf + boff + k;
            a = cat16(cvt8f(ap), cvt8f(ap + 16));
            b = cat16(cvt8f(bp), cvt8f(bp + 16));
        }
        acc = __builtin_amdgcn_wmma_f32_16x16x32_bf16(false, a, false, b,
                                                      (short)0, acc, false, false);
    }

    const int col   = colTile * 16 + l16;
    const float bias = bh[col];
    const int rbase = rowTile * 16 + hi * 8;
    __bf16* hb = (__bf16*)hidden;
#pragma unroll
    for (int j = 0; j < 8; ++j) {
        float v = acc[j] + bias;
        v = v > 0.f ? v : 0.f;
        hb[(size_t)(rbase + j) * N_HID + col] = (__bf16)v;
    }
}

// ---------------- Stage 2: per-head out = hidden_i @ W_i^T + b_i ----------------
// grid (2815, 64), block 256. Wave handles global col tile t = blockIdx.x*8+wave
// (0..22519); head found via prefix-sum scan. K = 128, 4 WMMA steps.
template <bool BF16W>
__global__ void __launch_bounds__(256)
stage2_kernel(const unsigned short* __restrict__ hidden,
              const float* __restrict__ Wf, const unsigned short* __restrict__ Wb,
              const float* __restrict__ bheads, float* __restrict__ out) {
    const int lane = threadIdx.x & 31;
    const int wave = threadIdx.x >> 5;
    const int l16  = lane & 15;
    const int hi   = lane >> 4;
    const int t       = blockIdx.x * 8 + wave;   // global col tile
    const int rowTile = blockIdx.y;

    int head = 0;
    while (head < NUM_HEADS - 1 && t >= c_cto[head + 1]) head++;
    const int lct  = t - c_cto[head];
    const int arow = rowTile * 16 + l16;
    const int ncol = lct * 16 + l16;             // out col, also W_heads row

    const __bf16* hb = (const __bf16*)hidden;
    const size_t abase = (size_t)arow * N_HID + head * D_SLICE + hi * 8;
    const size_t wbase = ((size_t)head * P_MAX + ncol) * D_SLICE + hi * 8;

    v8f acc = {};
#pragma unroll
    for (int k = 0; k < D_SLICE; k += 32) {
        v16bf a = cat16(*(const v8bf*)(hb + abase + k),
                        *(const v8bf*)(hb + abase + k + 16));
        v16bf b;
        if constexpr (BF16W) {
            const __bf16* bp = (const __bf16*)Wb + wbase + k;
            b = cat16(*(const v8bf*)bp, *(const v8bf*)(bp + 16));
        } else {
            const float* bp = Wf + wbase + k;
            b = cat16(cvt8f(bp), cvt8f(bp + 16));
        }
        acc = __builtin_amdgcn_wmma_f32_16x16x32_bf16(false, a, false, b,
                                                      (short)0, acc, false, false);
    }

    const float bias = bheads[(size_t)head * P_MAX + ncol];
    const int   p    = c_p[head];
    float* obase = out + (size_t)c_cto[head] * 16 * B_ROWS;
    const int rbase = rowTile * 16 + hi * 8;
#pragma unroll
    for (int j = 0; j < 8; ++j)
        obase[(size_t)(rbase + j) * p + ncol] = acc[j] + bias;
}

extern "C" void kernel_launch(void* const* d_in, const int* in_sizes, int n_in,
                              void* d_out, int out_size, void* d_ws, size_t ws_size,
                              hipStream_t stream) {
    const float* x      = (const float*)d_in[0];
    const float* Wh     = (const float*)d_in[1];
    const float* bh     = (const float*)d_in[2];
    const float* Whds   = (const float*)d_in[3];
    const float* bheads = (const float*)d_in[4];
    float* out = (float*)d_out;
    char*  ws  = (char*)d_ws;

    auto alignUp = [](size_t v) { return (v + 255) & ~(size_t)255; };
    const size_t hidBytes = (size_t)B_ROWS * N_HID * 2;               // 3.67 MB
    const size_t xbBytes  = (size_t)B_ROWS * D_IN * 2;                // 3.93 MB
    const size_t whBytes  = (size_t)N_HID * D_IN * 2;                 // 6.88 MB
    const size_t wdBytes  = (size_t)NUM_HEADS * P_MAX * D_SLICE * 2;  // 132 MB (< 192MB L2)

    size_t offHid = 0;
    size_t offXb  = alignUp(offHid + hidBytes);
    size_t offWh  = alignUp(offXb + xbBytes);
    size_t offWd  = alignUp(offWh + whBytes);
    size_t total  = alignUp(offWd + wdBytes);
    const bool full = ws_size >= total;

    unsigned short* hid = (unsigned short*)(ws + offHid);
    const dim3 blk(256);
    const dim3 g1(14, 64);     // 112 col tiles / 8 waves
    const dim3 g2(2815, 64);   // 22520 col tiles / 8 waves

    if (full) {
        unsigned short* xb  = (unsigned short*)(ws + offXb);
        unsigned short* whb = (unsigned short*)(ws + offWh);
        unsigned short* wdb = (unsigned short*)(ws + offWd);
        long n4x = (long)B_ROWS * D_IN / 4;
        long n4w = (long)N_HID * D_IN / 4;
        long n4d = (long)NUM_HEADS * P_MAX * D_SLICE / 4;
        cvt_f32_bf16_kernel<<<dim3((unsigned)((n4x + 255) / 256)), blk, 0, stream>>>(x, xb, n4x);
        cvt_f32_bf16_kernel<<<dim3((unsigned)((n4w + 255) / 256)), blk, 0, stream>>>(Wh, whb, n4w);
        cvt_f32_bf16_kernel<<<dim3((unsigned)((n4d + 255) / 256)), blk, 0, stream>>>(Whds, wdb, n4d);
        stage1_kernel<true><<<g1, blk, 0, stream>>>(nullptr, nullptr, xb, whb, bh, hid);
        stage2_kernel<true><<<g2, blk, 0, stream>>>(hid, nullptr, wdb, bheads, out);
    } else {
        stage1_kernel<false><<<g1, blk, 0, stream>>>(x, Wh, nullptr, nullptr, bh, hid);
        stage2_kernel<false><<<g2, blk, 0, stream>>>(hid, Whds, nullptr, bheads, out);
    }
}
